// BiLSTMClassifier_37658273252034
// MI455X (gfx1250) — compile-verified, matrix operand order checked by reading, not run on hardware
//
#include <hip/hip_runtime.h>
#include <hip/hip_bf16.h>

typedef __attribute__((ext_vector_type(16))) __bf16 v16bf;
typedef __attribute__((ext_vector_type(8)))  float  v8f;

#define TT 512
#define BB 128
#define HH 128
#define G4 512   // 4*H
// xp element count per direction (bf16): T*B*4H
#define XP_PER_DIR (512L * 128L * 512L)

__device__ __forceinline__ int kswz(int j, int khalf) {
    // 16-bit A/B fragment K index for element j of v16bf (ISA 7.12.2):
    // VGPR r<4 -> K = 2r+h (+khalf*8), VGPR r>=4 -> K = 16+2(r-4)+h (+khalf*8)
    return (j & 7) + ((j >> 3) << 4) + khalf;
}

__device__ __forceinline__ float sigmoidf_(float x) {
    return 1.0f / (1.0f + expf(-x));
}

// ---------------------------------------------------------------------------
// Kernel 1: fused embedding gather + input projection (both directions)
//   xp[dir][t*B+b][g] = sum_e emb[x[b, t or T-1-t]][e] * Wih[g][e] + bih[g]+bhh[g]
// One wave computes one 16x16 output tile, K=128 via 4x v_wmma_f32_16x16x32_bf16.
// ---------------------------------------------------------------------------
__global__ __launch_bounds__(256)
void bilstm_proj_kernel(const int* __restrict__ x,
                        const float* __restrict__ emb,
                        const float* __restrict__ Wih_f,
                        const float* __restrict__ Wih_b,
                        const float* __restrict__ bih_f,
                        const float* __restrict__ bhh_f,
                        const float* __restrict__ bih_b,
                        const float* __restrict__ bhh_b,
                        __bf16* __restrict__ xp)
{
    const int lane = threadIdx.x & 31;
    const int wave = threadIdx.x >> 5;
    const long gw = (long)blockIdx.x * 8 + wave;      // global wave id
    const int tilesPerDir = 4096 * 32;                // (B*T/16) * (4H/16)
    const int dir = (int)(gw / tilesPerDir);
    const int rem = (int)(gw % tilesPerDir);
    const int mt = rem >> 5;                          // 0..4095 M-tile
    const int nt = rem & 31;                          // 0..31   N-tile

    const float* Wih = dir ? Wih_b : Wih_f;
    const float* bi  = dir ? bih_b : bih_f;
    const float* bh  = dir ? bhh_b : bhh_f;

    // A row handled by this lane (rows split across lanes; lane[4] picks K half)
    const int mrow = mt * 16 + (lane & 15);           // flat row = t*B + b
    const int t = mrow >> 7;
    const int b = mrow & 127;
    const int tt = dir ? (TT - 1 - t) : t;
    const int tok = x[b * TT + tt];
    const float* arow = emb + (long)tok * 128;

    const int n = nt * 16 + (lane & 15);              // gate index for this lane
    const float* brow = Wih + n * 128;                // row n of Wih == col n of B
    const int khalf = (lane >> 4) << 3;

    v8f acc = {};
    #pragma unroll
    for (int kk = 0; kk < 4; ++kk) {
        const int k0 = kk * 32;
        v16bf av, bv;
        #pragma unroll
        for (int j = 0; j < 16; ++j) {
            const int k = k0 + kswz(j, khalf);
            av[j] = (__bf16)arow[k];
            bv[j] = (__bf16)brow[k];
        }
        acc = __builtin_amdgcn_wmma_f32_16x16x32_bf16(
                  false, av, false, bv, (short)0, acc, false, false);
    }

    const float bias = bi[n] + bh[n];
    __bf16* outp = xp + (long)dir * XP_PER_DIR;
    #pragma unroll
    for (int r = 0; r < 8; ++r) {
        const int m = mt * 16 + r + ((lane >> 4) << 3);
        outp[(long)m * G4 + n] = (__bf16)(acc[r] + bias);
    }
}

// ---------------------------------------------------------------------------
// Kernel 2: LSTM scan, fused with max-over-time pooling.
// Grid = 16 workgroups: dir(2) x batch-group(8 of 16 rows). 512 threads = 16 waves.
// Per step: g = xp[t] + h @ Whh^T via WMMA (M=16, N=512, K=128).
// The 16KB xp slab for step t+1 is double-buffered into LDS with
// global_load_async_to_lds_b128 (ASYNCcnt), hiding global latency behind the
// recurrent GEMM + gate math. Whh^T B-fragments are preloaded once into VGPRs.
// ---------------------------------------------------------------------------
#define HPAD 136   // padded h row stride (bf16) to avoid LDS bank conflicts

__global__ __launch_bounds__(512)
void bilstm_scan_kernel(const __bf16* __restrict__ xp,
                        const float* __restrict__ Whh_f,
                        const float* __restrict__ Whh_b,
                        float* __restrict__ pooled)
{
    __shared__ float  g_lds[16 * G4];           // 32 KB gate pre-activations
    __shared__ __bf16 h_lds[16 * HPAD];         // padded h tile
    __shared__ __bf16 xbuf[2][16 * G4];         // 2 x 16 KB async xp double-buffer

    const int tid  = threadIdx.x;
    const int lane = tid & 31;
    const int wave = tid >> 5;            // 0..15, owns n-tiles {2w, 2w+1}
    const int dir  = blockIdx.x >> 3;
    const int bg   = blockIdx.x & 7;
    const int m0   = bg * 16;             // batch rows m0..m0+15

    const float*  Whh = dir ? Whh_b : Whh_f;
    const __bf16* xpd = xp + (long)dir * XP_PER_DIR + (long)m0 * G4;
    const int khalf = (lane >> 4) << 3;

    // Preload loop-invariant B fragments of Whh^T: 2 n-tiles x 4 k-steps
    v16bf bfrag[2][4];
    #pragma unroll
    for (int ntl = 0; ntl < 2; ++ntl) {
        const int n = wave * 32 + ntl * 16 + (lane & 15);
        const float* brow = Whh + n * HH;     // row n of Whh == col n of B
        #pragma unroll
        for (int kk = 0; kk < 4; ++kk) {
            #pragma unroll
            for (int j = 0; j < 16; ++j)
                bfrag[ntl][kk][j] = (__bf16)brow[kk * 32 + kswz(j, khalf)];
        }
    }

    // h = 0
    for (int i = tid; i < 16 * HPAD; i += 512) h_lds[i] = (__bf16)0.0f;

    // per-lane async-copy addressing: 512 threads x 32B = 16KB slab
    const unsigned voff = (unsigned)tid * 32u;                  // global byte offset
    const unsigned ldsbase0 = (unsigned)(size_t)(&xbuf[0][0]);  // LDS offset (addr[31:0])
    const unsigned ldsbase1 = (unsigned)(size_t)(&xbuf[1][0]);

    // prefetch slab for t = 0 into xbuf[0]
    {
        const __bf16* gsrc = xpd;   // t = 0
        unsigned ldsAddr = ldsbase0 + voff;
        asm volatile("global_load_async_to_lds_b128 %0, %1, %2"
                     :: "v"(ldsAddr), "v"(voff), "s"(gsrc) : "memory");
    }

    // per-thread cell state + running max: 4 (b,u) elements per thread
    float c_st[4] = {0.f, 0.f, 0.f, 0.f};
    float mp[4]   = {-__builtin_inff(), -__builtin_inff(),
                     -__builtin_inff(), -__builtin_inff()};

    for (int t = 0; t < TT; ++t) {
        // (a) kick off async copy of next step's slab (clamped on last iter)
        {
            const int t1 = (t + 1 < TT) ? (t + 1) : t;
            const __bf16* gsrc = xpd + (long)t1 * (BB * G4);
            unsigned ldsAddr = (((t + 1) & 1) ? ldsbase1 : ldsbase0) + voff;
            asm volatile("global_load_async_to_lds_b128 %0, %1, %2"
                         :: "v"(ldsAddr), "v"(voff), "s"(gsrc) : "memory");
        }
        // (b) own slab for step t has landed (next-step copy stays in flight)
        asm volatile("s_wait_asynccnt 0x1" ::: "memory");
        // (c) all waves' slab pieces visible; h_lds from previous step ready
        __syncthreads();

        const __bf16* xs = &xbuf[t & 1][0];

        // accumulators init from input projection (LDS), then recurrent GEMM
        v8f acc[2];
        #pragma unroll
        for (int ntl = 0; ntl < 2; ++ntl) {
            const int n = wave * 32 + ntl * 16 + (lane & 15);
            #pragma unroll
            for (int r = 0; r < 8; ++r) {
                const int m = r + ((lane >> 4) << 3);
                acc[ntl][r] = (float)xs[m * G4 + n];
            }
        }
        #pragma unroll
        for (int kk = 0; kk < 4; ++kk) {
            v16bf av;
            const int m = lane & 15;
            #pragma unroll
            for (int j = 0; j < 16; ++j)
                av[j] = h_lds[m * HPAD + kk * 32 + kswz(j, khalf)];
            acc[0] = __builtin_amdgcn_wmma_f32_16x16x32_bf16(
                         false, av, false, bfrag[0][kk], (short)0, acc[0], false, false);
            acc[1] = __builtin_amdgcn_wmma_f32_16x16x32_bf16(
                         false, av, false, bfrag[1][kk], (short)0, acc[1], false, false);
        }
        // spill gate pre-activations to LDS
        #pragma unroll
        for (int ntl = 0; ntl < 2; ++ntl) {
            const int n = wave * 32 + ntl * 16 + (lane & 15);
            #pragma unroll
            for (int r = 0; r < 8; ++r) {
                const int m = r + ((lane >> 4) << 3);
                g_lds[m * G4 + n] = acc[ntl][r];
            }
        }
        __syncthreads();

        // gates: i,f,g,o (PyTorch order) -> c,h ; 4 elements per thread
        #pragma unroll
        for (int q = 0; q < 4; ++q) {
            const int e = tid * 4 + q;            // 0..2047
            const int m = e >> 7;                 // local batch row 0..15
            const int u = e & 127;                // hidden unit
            const float ig = sigmoidf_(g_lds[m * G4 + u]);
            const float fg = sigmoidf_(g_lds[m * G4 + 128 + u]);
            const float gg = tanhf(    g_lds[m * G4 + 256 + u]);
            const float og = sigmoidf_(g_lds[m * G4 + 384 + u]);
            const float c  = fg * c_st[q] + ig * gg;
            c_st[q] = c;
            const float h = og * tanhf(c);
            mp[q] = fmaxf(mp[q], h);
            h_lds[m * HPAD + u] = (__bf16)h;
        }
    }
    // drain remaining async op before exit
    asm volatile("s_wait_asynccnt 0x0" ::: "memory");

    // pooled: (B, 2H), forward -> cols [0,128), backward -> cols [128,256)
    #pragma unroll
    for (int q = 0; q < 4; ++q) {
        const int e = tid * 4 + q;
        const int m = e >> 7;
        const int u = e & 127;
        pooled[(m0 + m) * 256 + dir * 128 + u] = mp[q];
    }
}

// ---------------------------------------------------------------------------
// Kernel 3: tiny MLP head: relu(pooled @ W1^T + b1) @ W2^T + b2 -> sigmoid
// ---------------------------------------------------------------------------
__global__ __launch_bounds__(128)
void bilstm_mlp_kernel(const float* __restrict__ pooled,
                       const float* __restrict__ W1, const float* __restrict__ b1,
                       const float* __restrict__ W2, const float* __restrict__ b2,
                       float* __restrict__ out)
{
    const int b = threadIdx.x;
    const float* p = pooled + b * 256;
    float acc2 = b2[0];
    for (int j = 0; j < 64; ++j) {
        float s = b1[j];
        const float* w = W1 + j * 256;
        #pragma unroll 8
        for (int k = 0; k < 256; ++k) s += p[k] * w[k];
        acc2 += fmaxf(s, 0.0f) * W2[j];
    }
    out[b] = sigmoidf_(acc2);
}

// ---------------------------------------------------------------------------
extern "C" void kernel_launch(void* const* d_in, const int* in_sizes, int n_in,
                              void* d_out, int out_size, void* d_ws, size_t ws_size,
                              hipStream_t stream) {
    const int*   x      = (const int*)  d_in[0];
    const float* emb    = (const float*)d_in[1];
    const float* Wih_f  = (const float*)d_in[2];
    const float* Whh_f  = (const float*)d_in[3];
    const float* bih_f  = (const float*)d_in[4];
    const float* bhh_f  = (const float*)d_in[5];
    const float* Wih_b  = (const float*)d_in[6];
    const float* Whh_b  = (const float*)d_in[7];
    const float* bih_b  = (const float*)d_in[8];
    const float* bhh_b  = (const float*)d_in[9];
    const float* W1     = (const float*)d_in[10];
    const float* b1     = (const float*)d_in[11];
    const float* W2     = (const float*)d_in[12];
    const float* b2     = (const float*)d_in[13];
    float* out = (float*)d_out;

    // workspace: xp (bf16, 2 dirs) then pooled (f32, B x 2H)
    __bf16* xp     = (__bf16*)d_ws;
    float*  pooled = (float*)((char*)d_ws + 2 * XP_PER_DIR * sizeof(__bf16));

    // 1) embedding + input projection: 2*4096*32 tiles, 8 waves/block
    bilstm_proj_kernel<<<32768, 256, 0, stream>>>(
        x, emb, Wih_f, Wih_b, bih_f, bhh_f, bih_b, bhh_b, xp);

    // 2) batch-parallel bidirectional LSTM scan + max pooling (async xp pipeline)
    bilstm_scan_kernel<<<16, 512, 0, stream>>>(xp, Whh_f, Whh_b, pooled);

    // 3) classifier head
    bilstm_mlp_kernel<<<1, 128, 0, stream>>>(pooled, W1, b1, W2, b2, out);
}
